// Hyena_79568564126059
// MI455X (gfx1250) — compile-verified, matrix operand order checked by reading, not run on hardware
//
#include <hip/hip_runtime.h>
#include <hip/hip_bf16.h>
#include <math.h>

typedef __attribute__((ext_vector_type(16))) _Float16 v16h;
typedef __attribute__((ext_vector_type(8)))  float    v8f;
typedef unsigned int u32x4 __attribute__((ext_vector_type(4)));
typedef int          i32x4 __attribute__((ext_vector_type(4)));
typedef int          i32x8 __attribute__((ext_vector_type(8)));

#define LSEQ   8192
#define DMODEL 256
#define BATCH  2
#define NFFT   16384
#define MROWS  (BATCH * LSEQ)
#define PI_F   3.14159265358979323846f

__device__ __forceinline__ float gelu_exact(float x) {
  return 0.5f * x * (1.0f + erff(x * 0.70710678118654752f));
}

// ---------------------------------------------------------------------------
// CDNA5 async VMEM -> LDS (ASYNCcnt-tracked), bypasses VGPR staging.
// ---------------------------------------------------------------------------
__device__ __forceinline__ void async_gl2lds_b128(void* lds_dst, const void* gsrc) {
  unsigned loff = (unsigned)(unsigned long long)lds_dst;   // LDS byte address
  unsigned long long ga = (unsigned long long)gsrc;
  asm volatile("global_load_async_to_lds_b128 %0, %1, off"
               :: "v"(loff), "v"(ga) : "memory");
}
__device__ __forceinline__ void wait_asynccnt0() {
  asm volatile("s_wait_asynccnt 0x0" ::: "memory");
}

// ---------------------------------------------------------------------------
// CDNA5 Tensor Data Mover: 1-D contiguous tile descriptor (ISA ch.8).
// group0: [1:0]=count=1 | [63:32]=lds_addr | [120:64]=global_addr | type=2
// group1: data_size=2(4B); tensor_dim0=tile_dim0=ndw; tensor_dim1=tile_dim1=1;
//         tensor_dim0_stride=ndw. groups 2/3 zero (<=2-D tensor).
// ---------------------------------------------------------------------------
__device__ __forceinline__ void tdm_desc_1d(u32x4& g0, i32x8& g1,
    const void* lds_ptr, const void* gptr, unsigned ndw) {
  unsigned lds_addr = (unsigned)(unsigned long long)lds_ptr;
  unsigned long long ga = (unsigned long long)gptr;
  g0[0] = 1u;                                           // count=1 (user D#)
  g0[1] = lds_addr;
  g0[2] = (unsigned)ga;
  g0[3] = (unsigned)((ga >> 32) & 0x1FFFFFFull) | (2u << 30);  // addr[56:32]|type=2
  g1[0] = (int)(2u << 16);                              // data_size: 4 bytes
  g1[1] = (int)((ndw & 0xFFFFu) << 16);                 // tensor_dim0[15:0]
  g1[2] = (int)(((ndw >> 16) & 0xFFFFu) | (1u << 16));  // dim0[31:16] | tensor_dim1=1
  g1[3] = (int)(ndw << 16);                             // tile_dim0 (16-bit)
  g1[4] = 1;                                            // tile_dim1=1, tile_dim2=0
  g1[5] = (int)ndw;                                     // tensor_dim0_stride[31:0]
  g1[6] = 0;
  g1[7] = 0;
}

#if __has_builtin(__builtin_amdgcn_tensor_load_to_lds)
#define HYENA_HAS_TDM 1
__device__ __forceinline__ void tdm_load_1d(void* lds_dst, const void* gsrc, unsigned ndw) {
  u32x4 g0; i32x8 g1;
  tdm_desc_1d(g0, g1, lds_dst, gsrc, ndw);
  i32x4 z4 = {0, 0, 0, 0};
#if __clang_major__ >= 23
  i32x8 z8 = {0, 0, 0, 0, 0, 0, 0, 0};
  __builtin_amdgcn_tensor_load_to_lds(g0, g1, z4, z4, z8, 0);
#else
  __builtin_amdgcn_tensor_load_to_lds(g0, g1, z4, z4, 0);
#endif
}
__device__ __forceinline__ void tdm_store_1d(const void* lds_src, void* gdst, unsigned ndw) {
  u32x4 g0; i32x8 g1;
  tdm_desc_1d(g0, g1, lds_src, gdst, ndw);
  i32x4 z4 = {0, 0, 0, 0};
#if __clang_major__ >= 23
  i32x8 z8 = {0, 0, 0, 0, 0, 0, 0, 0};
  __builtin_amdgcn_tensor_store_from_lds(g0, g1, z4, z4, z8, 0);
#else
  __builtin_amdgcn_tensor_store_from_lds(g0, g1, z4, z4, 0);
#endif
}
#else
#define HYENA_HAS_TDM 0
#endif

// ---------------------------------------------------------------------------
// WMMA GEMM:  C = A[M,K](f16) * W[N,K]^T(f16) + bias
// Block tile 128x64, K-step 32, 8 waves: 4x2 wave grid, 32x32 per wave.
// Double-buffered LDS tiles filled by global_load_async_to_lds_b128 so the
// DMA of tile i+1 overlaps the WMMA of tile i.
// tmode==0: C[M,N] row-major.  tmode==1: C stored pre-transposed (B,N,LSEQ).
// ---------------------------------------------------------------------------
union FragH { v16h h; unsigned u[8]; };

__global__ __launch_bounds__(256) void gemm_f16_wmma(
    const _Float16* __restrict__ A, const _Float16* __restrict__ W,
    const float* __restrict__ bias, float* __restrict__ C,
    int M, int N, int K, int tmode)
{
  constexpr int BM = 128, BN = 64, BK = 32;
  __shared__ _Float16 As[2][BM][BK + 8];
  __shared__ _Float16 Ws[2][BN][BK + 8];
  const int tid  = threadIdx.x;
  const int lane = tid & 31;
  const int wave = tid >> 5;
  const int wm = wave & 3, wn = wave >> 2;
  const int m0 = blockIdx.x * BM, n0 = blockIdx.y * BN;
  const int mlane = lane & 15, half = lane >> 4;

  v8f acc[2][2];
#pragma unroll
  for (int i = 0; i < 2; ++i)
#pragma unroll
    for (int j = 0; j < 2; ++j)
#pragma unroll
      for (int r = 0; r < 8; ++r) acc[i][j][r] = 0.0f;

  const int arow = tid >> 1, achunk = tid & 1;     // 128 rows x 2 chunks
  auto stage = [&](int kb, int buf) {
    const _Float16* ga = A + (size_t)(m0 + arow) * K + kb + achunk * 16;
    async_gl2lds_b128(&As[buf][arow][achunk * 16],     ga);
    async_gl2lds_b128(&As[buf][arow][achunk * 16 + 8], ga + 8);
    if (tid < 128) {                                  // waves 0..3, no divergence
      const _Float16* gw = W + (size_t)(n0 + arow) * K + kb + achunk * 16;
      async_gl2lds_b128(&Ws[buf][arow][achunk * 16],     gw);
      async_gl2lds_b128(&Ws[buf][arow][achunk * 16 + 8], gw + 8);
    }
  };

  const int ntiles = K / BK;
  stage(0, 0);
  wait_asynccnt0();
  __syncthreads();

  for (int ti = 0; ti < ntiles; ++ti) {
    const int buf = ti & 1;
    if (ti + 1 < ntiles) stage((ti + 1) * BK, buf ^ 1);   // overlap DMA w/ math

#pragma unroll
    for (int mi = 0; mi < 2; ++mi) {
      FragH fa;
      const _Float16* ap = &As[buf][wm * 32 + mi * 16 + mlane][0];
#pragma unroll
      for (int v = 0; v < 8; ++v) {
        int kk = 2 * v + ((v >= 4) ? 8 : 0) + half * 8;
        fa.u[v] = *(const unsigned*)(ap + kk);
      }
#pragma unroll
      for (int ni = 0; ni < 2; ++ni) {
        FragH fb;
        const _Float16* bp = &Ws[buf][wn * 32 + ni * 16 + mlane][0];
#pragma unroll
        for (int v = 0; v < 8; ++v) {
          int kk = 2 * v + ((v >= 4) ? 8 : 0) + half * 8;
          fb.u[v] = *(const unsigned*)(bp + kk);
        }
        acc[mi][ni] = __builtin_amdgcn_wmma_f32_16x16x32_f16(
            false, fa.h, false, fb.h, (short)0, acc[mi][ni], false, false);
      }
    }
    wait_asynccnt0();
    __syncthreads();
  }

#pragma unroll
  for (int mi = 0; mi < 2; ++mi)
#pragma unroll
    for (int ni = 0; ni < 2; ++ni) {
      int crow = m0 + wm * 32 + mi * 16 + half * 8;
      int ccol = n0 + wn * 32 + ni * 16 + mlane;
      float bv = bias[ccol];
      if (!tmode) {
#pragma unroll
        for (int r = 0; r < 8; ++r)
          C[(size_t)(crow + r) * N + ccol] = acc[mi][ni][r] + bv;
      } else {
        int b = crow >> 13, t = crow & (LSEQ - 1);
        float* dst = C + ((size_t)b * N + ccol) * LSEQ + t;
#pragma unroll
        for (int r = 0; r < 8; ++r) dst[r] = acc[mi][ni][r] + bv;
      }
    }
}

// ---------------------------------------------------------------------------
// LayerNorm (one wave32 per row of 256) fused with f32->f16 conversion.
// ---------------------------------------------------------------------------
__global__ __launch_bounds__(256) void ln_to_f16(
    const float* __restrict__ X, const float* __restrict__ g,
    const float* __restrict__ b, _Float16* __restrict__ A16)
{
  const int wave = threadIdx.x >> 5, lane = threadIdx.x & 31;
  const int row = blockIdx.x * 8 + wave;
  const float* xr = X + (size_t)row * DMODEL + lane * 8;
  float v[8]; float s = 0.f;
  float4 a0 = ((const float4*)xr)[0], a1 = ((const float4*)xr)[1];
  v[0]=a0.x; v[1]=a0.y; v[2]=a0.z; v[3]=a0.w;
  v[4]=a1.x; v[5]=a1.y; v[6]=a1.z; v[7]=a1.w;
#pragma unroll
  for (int i = 0; i < 8; ++i) s += v[i];
#pragma unroll
  for (int o = 16; o >= 1; o >>= 1) s += __shfl_xor(s, o, 32);
  const float mean = s * (1.0f / 256.0f);
  float q = 0.f;
#pragma unroll
  for (int i = 0; i < 8; ++i) { float d = v[i] - mean; q += d * d; }
#pragma unroll
  for (int o = 16; o >= 1; o >>= 1) q += __shfl_xor(q, o, 32);
  const float r = rsqrtf(q * (1.0f / 256.0f) + 1e-5f);
  union { _Float16 h[8]; uint4 u4; } o16;
#pragma unroll
  for (int i = 0; i < 8; ++i) {
    int d = lane * 8 + i;
    o16.h[i] = (_Float16)((v[i] - mean) * r * g[d] + b[d]);
  }
  *(uint4*)(A16 + (size_t)row * DMODEL + lane * 8) = o16.u4;
}

// ---------------------------------------------------------------------------
// Hyena filter MLP: one thread per time-step t, emits k[d][t] for all 256 d.
// ---------------------------------------------------------------------------
__global__ __launch_bounds__(256) void hyena_filter_kernel(
    const float* __restrict__ w0, const float* __restrict__ b0,
    const float* __restrict__ w1, const float* __restrict__ b1,
    const float* __restrict__ w2, const float* __restrict__ b2,
    const float* __restrict__ w3, const float* __restrict__ freq,
    float* __restrict__ k)
{
  const int t = blockIdx.x * 256 + threadIdx.x;
  const float t01 = (float)t * (1.0f / (float)(LSEQ - 1));
  const float ang = (2.0f * PI_F / (float)LSEQ) * (float)t * 1.0e-4f;
  float sn, cs; sincosf(ang, &sn, &cs);
  const float z0 = t01, z1 = cs, z2 = -sn;
  float h[64], h2[64];
  for (int j = 0; j < 64; ++j)
    h[j] = sinf((w0[j * 3] * z0 + w0[j * 3 + 1] * z1 + w0[j * 3 + 2] * z2 + b0[j]) * freq[j]);
  for (int j = 0; j < 64; ++j) {
    float a = b1[j];
    for (int i = 0; i < 64; ++i) a += w1[j * 64 + i] * h[i];
    h2[j] = sinf(a * freq[j]);
  }
  for (int j = 0; j < 64; ++j) {
    float a = b2[j];
    for (int i = 0; i < 64; ++i) a += w2[j * 64 + i] * h2[i];
    h[j] = sinf(a * freq[j]);
  }
  const float dmin = -3.0701134573253944f;   // log(0.01)/1.5
  const float dmax = -15.350567286626972f;   // log(0.01)/0.3
  for (int d = 0; d < DMODEL; ++d) {
    float a = 0.f;
    for (int i = 0; i < 64; ++i) a += w3[d * 64 + i] * h[i];
    float delta = dmin + (dmax - dmin) * ((float)d * (1.0f / 255.0f));
    k[(size_t)d * LSEQ + t] = a * expf(-t01 * fabsf(delta));
  }
}

// ---------------------------------------------------------------------------
// 16384-pt complex FFT in LDS (128KB dynamic LDS; CDNA5 WGP has 320KB).
// Forward DIF (natural -> bit-rev), inverse DIT (bit-rev -> natural);
// pointwise products in bit-reversed order (K-hat stored the same way).
// ---------------------------------------------------------------------------
__device__ __forceinline__ void fft_dif16k(float* re, float* im, int tid) {
  __syncthreads();
#pragma unroll 1
  for (int s = 13; s >= 0; --s) {
    const int m = 1 << s;
    const float invm = 1.0f / (float)m;
    for (int idx = tid; idx < (NFFT >> 1); idx += 256) {
      const int j  = idx & (m - 1);
      const int i0 = ((idx >> s) << (s + 1)) + j;
      const int i1 = i0 + m;
      float ar = re[i0], ai = im[i0], br = re[i1], bi = im[i1];
      float sn, cs; sincosf(-PI_F * (float)j * invm, &sn, &cs);
      re[i0] = ar + br; im[i0] = ai + bi;
      float dr = ar - br, di = ai - bi;
      re[i1] = dr * cs - di * sn;
      im[i1] = dr * sn + di * cs;
    }
    __syncthreads();
  }
}

__device__ __forceinline__ void ifft_dit16k(float* re, float* im, int tid) {
  __syncthreads();
#pragma unroll 1
  for (int s = 0; s <= 13; ++s) {
    const int m = 1 << s;
    const float invm = 1.0f / (float)m;
    for (int idx = tid; idx < (NFFT >> 1); idx += 256) {
      const int j  = idx & (m - 1);
      const int i0 = ((idx >> s) << (s + 1)) + j;
      const int i1 = i0 + m;
      float ar = re[i0], ai = im[i0], br = re[i1], bi = im[i1];
      float sn, cs; sincosf(PI_F * (float)j * invm, &sn, &cs);
      float tr = br * cs - bi * sn;
      float ti = br * sn + bi * cs;
      re[i0] = ar + tr; im[i0] = ai + ti;
      re[i1] = ar - tr; im[i1] = ai - ti;
    }
    __syncthreads();
  }
}

// Fill re[0..8191] from global (TDM single-descriptor DMA when available,
// else async-b128), zero re[8192..] and im[] with all 256 threads.
__device__ __forceinline__ void load_row_lds(
    float* re, float* im, const float* __restrict__ src, int tid)
{
  float4 z4 = make_float4(0.f, 0.f, 0.f, 0.f);
#pragma unroll
  for (int j = 0; j < 8; ++j) ((float4*)&re[LSEQ])[tid + j * 256] = z4;
#pragma unroll
  for (int j = 0; j < 16; ++j) ((float4*)im)[tid + j * 256] = z4;
#if HYENA_HAS_TDM
  if (tid < 32) {                        // wave 0 issues one TDM descriptor
    tdm_load_1d(re, src, LSEQ);
    __builtin_amdgcn_s_wait_tensorcnt(0);
  }
#else
#pragma unroll
  for (int j = 0; j < 8; ++j) {
    int c = tid + j * 256;               // 16B chunk index (2048 total)
    async_gl2lds_b128(&re[c * 4], src + c * 4);
  }
  wait_asynccnt0();
#endif
}

__global__ __launch_bounds__(256) void fft_k_kernel(
    const float* __restrict__ k, float* __restrict__ Khre, float* __restrict__ Khim)
{
  extern __shared__ float smem[];
  float* re = smem; float* im = smem + NFFT;
  const int tid = threadIdx.x, d = blockIdx.x;
  load_row_lds(re, im, k + (size_t)d * LSEQ, tid);
  fft_dif16k(re, im, tid);
#if HYENA_HAS_TDM
  if (tid < 32) {                        // two TDM stores: whole spectra
    tdm_store_1d(re, Khre + (size_t)d * NFFT, NFFT);
    tdm_store_1d(im, Khim + (size_t)d * NFFT, NFFT);
    __builtin_amdgcn_s_wait_tensorcnt(0);
  }
#else
  for (int i = tid; i < NFFT; i += 256) {
    Khre[(size_t)d * NFFT + i] = re[i];
    Khim[(size_t)d * NFFT + i] = im[i];
  }
#endif
}

// One workgroup per (b,d) channel: full gated FFT convolution in LDS.
__global__ __launch_bounds__(256) void fftconv_kernel(
    const float* __restrict__ vx2, const float* __restrict__ x1,
    const float* __restrict__ Khre, const float* __restrict__ Khim,
    const float* __restrict__ fbias, float* __restrict__ outp)
{
  extern __shared__ float smem[];
  float* re = smem; float* im = smem + NFFT;
  const int tid = threadIdx.x;
  const int bd = blockIdx.x;            // b*256 + d
  const int d  = bd & (DMODEL - 1);
  const size_t base = (size_t)bd * LSEQ;

  load_row_lds(re, im, vx2 + base, tid);
  fft_dif16k(re, im, tid);
  const size_t kb = (size_t)d * NFFT;
  for (int i = tid; i < NFFT; i += 256) {
    float a = re[i], b = im[i];
    float c = Khre[kb + i], e = Khim[kb + i];
    re[i] = a * c - b * e;
    im[i] = a * e + b * c;
  }
  ifft_dit16k(re, im, tid);
  const float inv = 1.0f / (float)NFFT;
  const float fb  = fbias[d];
  for (int t = tid; t < LSEQ; t += 256) {
    float u = vx2[base + t];
    outp[base + t] = (re[t] * inv + u * fb) * x1[base + t];
  }
}

// ---------------------------------------------------------------------------
// Causal width-3 depthwise conv + split + gate. u is (B, 3C, L): contiguous t.
// ---------------------------------------------------------------------------
__global__ __launch_bounds__(256) void gate_kernel(
    const float* __restrict__ u, const float* __restrict__ sfw,
    const float* __restrict__ sfb, float* __restrict__ x1p, float* __restrict__ vx2p)
{
  const int bid = blockIdx.x;
  const int tc = bid & 31;              // L/256 chunks
  const int bd = bid >> 5;              // b*256 + d
  const int d  = bd & (DMODEL - 1);
  const int b  = bd >> 8;
  const int t  = tc * 256 + threadIdx.x;

  float uc[3];
#pragma unroll
  for (int s = 0; s < 3; ++s) {
    const int c = s * DMODEL + d;
    const size_t cb = ((size_t)b * 768 + c) * LSEQ + t;
    float a = sfb[c] + sfw[c * 3 + 2] * u[cb];
    if (t >= 1) a += sfw[c * 3 + 1] * u[cb - 1];
    if (t >= 2) a += sfw[c * 3 + 0] * u[cb - 2];
    uc[s] = a;
  }
  const size_t o = (size_t)bd * LSEQ + t;
  x1p[o]  = uc[0];
  vx2p[o] = uc[2] * uc[1];
}

// (B,D,L) f32 -> (B*L, D) f16 transpose for the output GEMM's A operand.
__global__ __launch_bounds__(256) void transpose_to_f16(
    const float* __restrict__ inp, _Float16* __restrict__ outp)
{
  __shared__ float tile[32][33];
  const int tx = threadIdx.x & 31, ty = threadIdx.x >> 5;
  const int t0 = blockIdx.x * 32, d0 = blockIdx.y * 32, b = blockIdx.z;
#pragma unroll
  for (int i = 0; i < 4; ++i) {
    int d = d0 + ty + i * 8;
    tile[ty + i * 8][tx] = inp[(size_t)(b * DMODEL + d) * LSEQ + t0 + tx];
  }
  __syncthreads();
#pragma unroll
  for (int i = 0; i < 4; ++i) {
    int t = t0 + ty + i * 8;
    outp[(size_t)(b * LSEQ + t) * DMODEL + d0 + tx] = (_Float16)tile[tx][ty + i * 8];
  }
}

// width-3 same-pad depthwise conv + gelu(gelu(.)) + residual, in-place into X.
__global__ __launch_bounds__(256) void postconv_kernel(
    const float* __restrict__ z, const float* __restrict__ cw, float* __restrict__ X)
{
  const int row = blockIdx.x;           // b*L + t
  const int d   = threadIdx.x;
  const int t   = row & (LSEQ - 1);
  float zm = (t >= 1)        ? z[(size_t)(row - 1) * DMODEL + d] : 0.f;
  float zc =                   z[(size_t)row       * DMODEL + d];
  float zp = (t < LSEQ - 1)  ? z[(size_t)(row + 1) * DMODEL + d] : 0.f;
  float y = cw[d * 3] * zm + cw[d * 3 + 1] * zc + cw[d * 3 + 2] * zp;
  y = gelu_exact(gelu_exact(y));
  X[(size_t)row * DMODEL + d] += y;
}

__global__ __launch_bounds__(256) void f32_to_f16_kernel(
    const float* __restrict__ a, _Float16* __restrict__ b, int n)
{
  int i = blockIdx.x * 256 + threadIdx.x;
  if (i < n) b[i] = (_Float16)a[i];
}

// ---------------------------------------------------------------------------
extern "C" void kernel_launch(void* const* d_in, const int* in_sizes, int n_in,
                              void* d_out, int out_size, void* d_ws, size_t ws_size,
                              hipStream_t stream) {
  (void)in_sizes; (void)n_in; (void)out_size; (void)ws_size;
  const float* x      = (const float*)d_in[0];
  const float* ln_g   = (const float*)d_in[1];
  const float* ln_b   = (const float*)d_in[2];
  const float* in_w   = (const float*)d_in[3];
  const float* in_b   = (const float*)d_in[4];
  const float* sf_w   = (const float*)d_in[5];
  const float* sf_b   = (const float*)d_in[6];
  const float* mlp_w0 = (const float*)d_in[7];
  const float* mlp_b0 = (const float*)d_in[8];
  const float* mlp_w1 = (const float*)d_in[9];
  const float* mlp_b1 = (const float*)d_in[10];
  const float* mlp_w2 = (const float*)d_in[11];
  const float* mlp_b2 = (const float*)d_in[12];
  const float* mlp_w3 = (const float*)d_in[13];
  const float* sfreq  = (const float*)d_in[14];
  const float* fbias  = (const float*)d_in[15];
  const float* out_w  = (const float*)d_in[16];
  const float* out_b  = (const float*)d_in[17];
  const float* conv_w = (const float*)d_in[18];

  char* ws = (char*)d_ws;
  size_t off = 0;
  auto bump = [&](size_t bytes) -> char* {
    off = (off + 255) & ~(size_t)255;
    char* p = ws + off;
    off += bytes;
    return p;
  };
  float*    kbuf  = (float*)bump((size_t)DMODEL * LSEQ * 4);
  float*    Khre  = (float*)bump((size_t)DMODEL * NFFT * 4);
  float*    Khim  = (float*)bump((size_t)DMODEL * NFFT * 4);
  _Float16* winh  = (_Float16*)bump((size_t)768 * 256 * 2);
  _Float16* wouth = (_Float16*)bump((size_t)256 * 256 * 2);
  float*    X     = (float*)bump((size_t)MROWS * DMODEL * 4);
  _Float16* A16   = (_Float16*)bump((size_t)MROWS * DMODEL * 2);
  float*    u     = (float*)bump((size_t)MROWS * 768 * 4);   // (B, 768, L)
  float*    x1p   = (float*)bump((size_t)BATCH * DMODEL * LSEQ * 4);
  float*    vx2p  = (float*)bump((size_t)BATCH * DMODEL * LSEQ * 4);
  float*    z     = (float*)bump((size_t)MROWS * DMODEL * 4);
  // u is dead after gate_kernel within each block iteration -> alias into it
  float*    convp = u;
  _Float16* A216  = (_Float16*)((char*)u + (size_t)BATCH * DMODEL * LSEQ * 4);

  // allow 128KB dynamic LDS (CDNA5 WGP supports up to 320KB)
  hipFuncSetAttribute((const void*)fft_k_kernel,
                      hipFuncAttributeMaxDynamicSharedMemorySize, 2 * NFFT * 4);
  hipFuncSetAttribute((const void*)fftconv_kernel,
                      hipFuncAttributeMaxDynamicSharedMemorySize, 2 * NFFT * 4);

  hipMemcpyAsync(X, x, (size_t)MROWS * DMODEL * 4, hipMemcpyDeviceToDevice, stream);
  f32_to_f16_kernel<<<768, 256, 0, stream>>>(in_w, winh, 768 * 256);
  f32_to_f16_kernel<<<256, 256, 0, stream>>>(out_w, wouth, 256 * 256);
  hyena_filter_kernel<<<LSEQ / 256, 256, 0, stream>>>(
      mlp_w0, mlp_b0, mlp_w1, mlp_b1, mlp_w2, mlp_b2, mlp_w3, sfreq, kbuf);
  fft_k_kernel<<<DMODEL, 256, 2 * NFFT * 4, stream>>>(kbuf, Khre, Khim);

  for (int it = 0; it < 12; ++it) {
    ln_to_f16<<<MROWS / 8, 256, 0, stream>>>(X, ln_g, ln_b, A16);
    gemm_f16_wmma<<<dim3(MROWS / 128, 768 / 64), 256, 0, stream>>>(
        A16, winh, in_b, u, MROWS, 768, 256, /*tmode=*/1);
    gate_kernel<<<BATCH * DMODEL * (LSEQ / 256), 256, 0, stream>>>(
        u, sf_w, sf_b, x1p, vx2p);
    fftconv_kernel<<<BATCH * DMODEL, 256, 2 * NFFT * 4, stream>>>(
        vx2p, x1p, Khre, Khim, fbias, convp);
    transpose_to_f16<<<dim3(LSEQ / 32, DMODEL / 32, BATCH), 256, 0, stream>>>(convp, A216);
    gemm_f16_wmma<<<dim3(MROWS / 128, DMODEL / 64), 256, 0, stream>>>(
        A216, wouth, out_b, z, MROWS, DMODEL, 256, /*tmode=*/0);
    postconv_kernel<<<MROWS, 256, 0, stream>>>(z, conv_w, X);
  }
  hipMemcpyAsync(d_out, X, (size_t)MROWS * DMODEL * 4, hipMemcpyDeviceToDevice, stream);
}
